// TopkRouter_4913442586644
// MI455X (gfx1250) — matrix operand order repaired and checked by reading, not compiled, not run on hardware
//
#include <hip/hip_runtime.h>

// ---------------------------------------------------------------------------
// MoE Top-K router, fused for gfx1250 (MI455X).
//   Pre-pass: W (f32) -> bf16, fragment-major WMMA-B layout in workspace.
//   Main:    logits = x @ W^T + b  via v_wmma_f32_16x16x32_bf16 (f32 accum),
//            top-8 on (logits + bi), softmax of unbiased logits * mask,
//            expert histogram via LDS + global atomics.
//   Epilogue: bi_new = bi + lr * sign(tokens/E - count_i)
// ---------------------------------------------------------------------------

typedef __attribute__((ext_vector_type(16))) __bf16 v16bf;
typedef __attribute__((ext_vector_type(8)))  float  v8f;

#define DIM        4096
#define NEXP       64
#define TPB        128      // tokens per block (4 waves x 32 tokens)
#define NWAVE      4
#define TOPKK      8
#define LDS_STRIDE 65       // 64 experts + 1 pad -> conflict-free column reads
#define FILTER_R   0.62f
#define LOAD_LR    0.001f

// Workspace layout: [0,256) bytes: unsigned counts[64]
//                   [256, 256 + 64*4096*2): bf16 W fragments
#define WB_OFFSET_BYTES 256
// Fragment-major: wb[kc][nt][lane][16], kc = k/32 (128 chunks), nt 0..3.
// Lane (lhalf*16 + l16) holds expert e = nt*16 + l16,
// K values kc*32 + lhalf*16 + (0..15)  == contiguous k order, same as the A fragments.

__device__ __forceinline__ void cvt8_to_bf16(v16bf& d, int base, float4 a, float4 b) {
    d[base + 0] = (__bf16)a.x; d[base + 1] = (__bf16)a.y;
    d[base + 2] = (__bf16)a.z; d[base + 3] = (__bf16)a.w;
    d[base + 4] = (__bf16)b.x; d[base + 5] = (__bf16)b.y;
    d[base + 6] = (__bf16)b.z; d[base + 7] = (__bf16)b.w;
}

__global__ void zero_counts_kernel(unsigned* __restrict__ counts) {
    counts[threadIdx.x] = 0u;
}

// One thread per (kc, nt, lane): 16384 threads, each converts 16 f32 -> bf16.
__global__ __launch_bounds__(256)
void convert_W_kernel(const float* __restrict__ W, __bf16* __restrict__ wb)
{
    const int gid   = blockIdx.x * 256 + threadIdx.x;   // [0, 16384)
    const int kc    = gid >> 7;                         // 128 k-chunks
    const int rem   = gid & 127;
    const int nt    = rem >> 5;                         // 4 n-tiles
    const int lane  = rem & 31;
    const int l16   = lane & 15;
    const int lhalf = lane >> 4;

    const float* src = W + (size_t)(nt * 16 + l16) * DIM + kc * 32 + lhalf * 16;
    const float4* p  = (const float4*)src;
    float4 g0 = p[0], g1 = p[1], g2 = p[2], g3 = p[3];

    v16bf frag;
    cvt8_to_bf16(frag, 0, g0, g1);
    cvt8_to_bf16(frag, 8, g2, g3);
    *(v16bf*)(wb + (size_t)gid * 16) = frag;            // 32B aligned store
}

__global__ __launch_bounds__(TPB)
void router_fused_kernel(const float* __restrict__ x,
                         const __bf16* __restrict__ wb,
                         const float* __restrict__ b,
                         const float* __restrict__ bi,
                         const float* __restrict__ rand_u,
                         float* __restrict__ out_router,
                         float* __restrict__ out_idx,
                         unsigned* __restrict__ counts)
{
    __shared__ float    lds[TPB * LDS_STRIDE];   // per-token logits (+b)
    __shared__ float    bi_lds[NEXP];
    __shared__ unsigned cnt[NEXP];

    const int tid = threadIdx.x;
    if (tid < NEXP) { bi_lds[tid] = bi[tid]; cnt[tid] = 0u; }

    const int wave  = tid >> 5;        // 0..3
    const int lane  = tid & 31;
    const int l16   = lane & 15;
    const int lhalf = lane >> 4;       // 0 or 1

    const int tokBlock = blockIdx.x * TPB;
    const int tok0     = tokBlock + wave * 32;   // this wave: 2 M-tiles of 16

    // A fragment pointers: lane holds row M=l16; half-lane picks K sub-range.
    const float* aptr[2];
    aptr[0] = x + (size_t)(tok0 +  0 + l16) * DIM + lhalf * 16;
    aptr[1] = x + (size_t)(tok0 + 16 + l16) * DIM + lhalf * 16;

    // Per-lane bias for this lane's expert columns.
    float bias_e[4];
#pragma unroll
    for (int nt = 0; nt < 4; ++nt) bias_e[nt] = b[nt * 16 + l16];

    v8f acc[2][4];
#pragma unroll
    for (int mt = 0; mt < 2; ++mt)
#pragma unroll
        for (int nt = 0; nt < 4; ++nt) { v8f z = {}; acc[mt][nt] = z; }

    // ---------------- GEMM main loop: K in chunks of 32 --------------------
    const __bf16* wlane = wb + (size_t)lane * 16;   // + (kc*4 + nt)*32*16
    for (int kc = 0; kc < DIM / 32; ++kc) {
        const int k = kc * 32;

        v16bf amat[2];
#pragma unroll
        for (int mt = 0; mt < 2; ++mt) {
            const float4* p = (const float4*)(aptr[mt] + k);
            float4 f0 = p[0], f1 = p[1];   // K: koff + 0..7
            float4 f2 = p[2], f3 = p[3];   // K: koff + 8..15
            cvt8_to_bf16(amat[mt], 0, f0, f1);
            cvt8_to_bf16(amat[mt], 8, f2, f3);
        }

        v16bf bmat[4];
#pragma unroll
        for (int nt = 0; nt < 4; ++nt)
            bmat[nt] = *(const v16bf*)(wlane + ((size_t)kc * 4 + nt) * 512);

#pragma unroll
        for (int mt = 0; mt < 2; ++mt)
#pragma unroll
            for (int nt = 0; nt < 4; ++nt)
                acc[mt][nt] = __builtin_amdgcn_wmma_f32_16x16x32_bf16(
                    /*neg_a=*/false, amat[mt],
                    /*neg_b=*/false, bmat[nt],
                    /*c_mod=*/(short)0, acc[mt][nt],
                    /*reuse_a=*/false, /*reuse_b=*/false);

        // Prefetch next A chunk (speculative; dropped if OOB).
        __builtin_prefetch(aptr[0] + k + 32, 0, 0);
        __builtin_prefetch(aptr[1] + k + 32, 0, 0);
    }

    // Spill logits (+ linear bias b) to LDS: [token_in_block][expert]
#pragma unroll
    for (int mt = 0; mt < 2; ++mt)
#pragma unroll
        for (int nt = 0; nt < 4; ++nt)
#pragma unroll
            for (int r = 0; r < 8; ++r) {
                int trow = wave * 32 + mt * 16 + lhalf * 8 + r;
                lds[trow * LDS_STRIDE + nt * 16 + l16] = acc[mt][nt][r] + bias_e[nt];
            }
    __syncthreads();

    // ---------------- Per-token top-8 / softmax / mask ---------------------
    {
        const int token = tokBlock + tid;          // one token per thread
        const float* row = &lds[tid * LDS_STRIDE];

        unsigned taken0 = 0u, taken1 = 0u;
        float vals[TOPKK];
        int   sel[TOPKK];
#pragma unroll
        for (int j = 0; j < TOPKK; ++j) {
            float best = -3.4e38f;
            int   bsel = 0;
            for (int e = 0; e < NEXP; ++e) {
                unsigned bit = (e < 32) ? ((taken0 >> e) & 1u)
                                        : ((taken1 >> (e - 32)) & 1u);
                float v = row[e] + bi_lds[e];      // biased selection score
                if (!bit && v > best) { best = v; bsel = e; }
            }
            if (bsel < 32) taken0 |= 1u << bsel; else taken1 |= 1u << (bsel - 32);
            sel[j]  = bsel;
            vals[j] = row[bsel];                   // un-biased logit for softmax
        }

        float m = vals[0];
#pragma unroll
        for (int j = 1; j < TOPKK; ++j) m = fmaxf(m, vals[j]);
        float ex[TOPKK];
        float s = 0.0f;
#pragma unroll
        for (int j = 0; j < TOPKK; ++j) { ex[j] = __expf(vals[j] - m); s += ex[j]; }
        float invs = 1.0f / s;

#pragma unroll
        for (int j = 0; j < TOPKK; ++j) {
            size_t o   = (size_t)token * TOPKK + j;
            float msk  = (rand_u[o] > FILTER_R) ? 1.0f : 0.0f;
            out_router[o] = ex[j] * invs * msk;
            out_idx[o]    = (float)sel[j];
            atomicAdd(&cnt[sel[j]], 1u);
        }
    }
    __syncthreads();
    if (tid < NEXP) atomicAdd(&counts[tid], cnt[tid]);
}

__global__ void bias_update_kernel(const float* __restrict__ bi,
                                   const unsigned* __restrict__ counts,
                                   float* __restrict__ out_bi,
                                   float c_avg)
{
    int e = threadIdx.x;
    float ei = c_avg - (float)counts[e];
    float sg = (ei > 0.0f ? 1.0f : 0.0f) - (ei < 0.0f ? 1.0f : 0.0f);
    out_bi[e] = bi[e] + LOAD_LR * sg;
}

extern "C" void kernel_launch(void* const* d_in, const int* in_sizes, int n_in,
                              void* d_out, int out_size, void* d_ws, size_t ws_size,
                              hipStream_t stream) {
    const float* x      = (const float*)d_in[0];   // [tokens, 4096]
    const float* W      = (const float*)d_in[1];   // [64, 4096]
    const float* b      = (const float*)d_in[2];   // [64]
    const float* bi     = (const float*)d_in[3];   // [64]
    const float* rand_u = (const float*)d_in[4];   // [tokens, 8]

    const int ntok = in_sizes[0] / DIM;            // 16384

    float*    out        = (float*)d_out;
    float*    out_router = out;                    // [tokens*8]
    float*    out_idx    = out + (size_t)ntok * TOPKK;       // [tokens*8]
    float*    out_bi     = out + (size_t)ntok * TOPKK * 2;   // [64]
    unsigned* counts     = (unsigned*)d_ws;        // [64] scratch histogram
    __bf16*   wb         = (__bf16*)((char*)d_ws + WB_OFFSET_BYTES);

    zero_counts_kernel<<<1, NEXP, 0, stream>>>(counts);
    convert_W_kernel<<<NEXP, 256, 0, stream>>>(W, wb);
    router_fused_kernel<<<ntok / TPB, TPB, 0, stream>>>(
        x, wb, b, bi, rand_u, out_router, out_idx, counts);
    bias_update_kernel<<<1, NEXP, 0, stream>>>(
        bi, counts, out_bi, (float)ntok / (float)NEXP);
}
